// SelfAtt_67637144978383
// MI455X (gfx1250) — matrix operand order repaired and checked by reading, not compiled, not run on hardware
//
#include <hip/hip_runtime.h>

typedef __attribute__((ext_vector_type(16))) _Float16 v16h;
typedef __attribute__((ext_vector_type(8)))  float    v8f;
typedef __attribute__((ext_vector_type(2)))  unsigned short u16x2;

namespace {
constexpr int kN     = 2048;   // sequence length
constexpr int kD     = 64;     // head dim
constexpr int kMTile = 128;    // query rows per workgroup (8 waves x 16)
constexpr int kKT    = 32;     // keys per inner iteration (WMMA K for S@V)
constexpr int kQStr  = 72;     // padded LDS row strides (f16 elems)
constexpr int kKStr  = 72;
constexpr int kVStr  = 36;
}

union F16Frag { unsigned int u[8]; _Float16 h[16]; v16h v; };
union U32H2   { unsigned int u; u16x2 h; };

// Native CDNA5 tanh transcendental (v_tanh_f32).
__device__ __forceinline__ float hw_tanhf(float x) {
#if __has_builtin(__builtin_amdgcn_tanhf)
  return __builtin_amdgcn_tanhf(x);
#elif __has_builtin(__builtin_amdgcn_tanh_f32)
  return __builtin_amdgcn_tanh_f32(x);
#else
  float r;
  asm("v_tanh_f32 %0, %1\n\tv_nop" : "=v"(r) : "v"(x));
  return r;
#endif
}

// Async memory->LDS 16-byte copy (GVS mode), tracked on ASYNCcnt.
__device__ __forceinline__ void async_copy_b128(unsigned lds_byte_addr,
                                                unsigned gmem_off,
                                                const void* gmem_base) {
  asm volatile("global_load_async_to_lds_b128 %0, %1, %2 offset:0"
               :: "v"(lds_byte_addr), "v"(gmem_off), "s"(gmem_base)
               : "memory");
}

__device__ __forceinline__ void wait_async0() {
#if __has_builtin(__builtin_amdgcn_s_wait_asynccnt)
  __builtin_amdgcn_s_wait_asynccnt(0);
#else
  asm volatile("s_wait_asynccnt 0x0" ::: "memory");
#endif
}

// Zero the f16 halves of `pair` whose mask byte (0/1) says "dropped".
// bytes (b_lo,b_hi) live at [sel] positions of mdw; perm expands them into
// the two 16-bit lanes, pk_sub turns 0->0xFFFF / 1->0x0000 (keep-mask).
__device__ __forceinline__ unsigned mask_pair(unsigned pair, unsigned mdw,
                                              unsigned sel) {
  U32H2 t;
  t.u = __builtin_amdgcn_perm(mdw, mdw, sel);
  t.h = t.h - (unsigned short)1;
  return pair & t.u;
}

__global__ __launch_bounds__(256) void attn_tanh_wmma(
    const float* __restrict__ Q, const float* __restrict__ K,
    const float* __restrict__ V, const unsigned char* __restrict__ Msk,
    float* __restrict__ O)
{
  __shared__ __align__(16) _Float16      Qs[kMTile * kQStr];
  __shared__ __align__(16) _Float16      Ks[kKT * kKStr];
  __shared__ __align__(16) _Float16      Vs[kD * kVStr];      // V transposed: [d][m]
  __shared__ __align__(16) unsigned char Ms[kMTile * kKT];

  const int bh    = blockIdx.y;
  const int qBase = blockIdx.x * kMTile;
  const int tid   = threadIdx.x;
  const int lane  = tid & 31;
  const int wave  = tid >> 5;
  const int lh    = lane >> 4;   // lane half (0/1)
  const int lr    = lane & 15;   // lane row within half

  const float*          Qg = Q + ((size_t)bh * kN + qBase) * kD;
  const float*          Kg = K + (size_t)bh * kN * kD;
  const float*          Vg = V + (size_t)bh * kN * kD;
  const unsigned char*  Mg = Msk + ((size_t)bh * kN + qBase) * kN;
  float*                Og = O + ((size_t)bh * kN + qBase) * kD;

  // K-index (pair base) maps per the ISA VGPR layout tables (16-bit data).
  int kmapA[8], kmapB[8];
#pragma unroll
  for (int v = 0; v < 8; ++v) {
    kmapA[v] = ((v < 4) ? (2 * v) : (16 + 2 * (v - 4))) + lh * 8;
    kmapB[v] = ((v < 4) ? (2 * v) : (16 + 2 * (v - 4))) + lh * 8;
  }

  // ---- Stage Q tile -> LDS f16, folding in the 1/sqrt(D) scale ----
#pragma unroll
  for (int i = tid; i < (kMTile * kD) / 4; i += 256) {
    int idx = i * 4;
    int row = idx / kD, d = idx % kD;
    float4 f = *(const float4*)(Qg + (size_t)row * kD + d);
    _Float16* p = &Qs[row * kQStr + d];
    p[0] = (_Float16)(f.x * 0.125f);
    p[1] = (_Float16)(f.y * 0.125f);
    p[2] = (_Float16)(f.z * 0.125f);
    p[3] = (_Float16)(f.w * 0.125f);
  }
  __syncthreads();

  // Q as loop-invariant B-fragments of the S^T = K*Q^T GEMM (lane = query col).
  F16Frag qb[2];
#pragma unroll
  for (int c = 0; c < 2; ++c)
#pragma unroll
    for (int v = 0; v < 8; ++v)
      qb[c].u[v] = *(const unsigned int*)&Qs[(wave * 16 + lr) * kQStr + c * 32 + kmapB[v]];

  v8f oacc[4] = {};   // O accumulators: 4 d-tiles of 16 cols, f32

  // Mask tile staging constants (one async b128 per thread per iteration).
  const unsigned mRow = (unsigned)(tid >> 1);
  const unsigned mOff = (unsigned)((tid & 1) * 16);
  const unsigned mLds = (unsigned)(size_t)&Ms[mRow * kKT + mOff];
  // This lane's mask row (its query row), two 8-byte runs per 32-key tile.
  const unsigned char* mRowPtr = &Ms[(wave * 16 + lr) * kKT];

#pragma unroll 1
  for (int kt = 0; kt < kN; kt += kKT) {
    // ---- Mask tile: 128 rows x 32 B, direct async memory->LDS copy ----
    async_copy_b128(mLds, mRow * (unsigned)kN + (unsigned)kt + mOff, Mg);

    // ---- Stage K tile (row-major f16) ----
#pragma unroll
    for (int i = tid; i < (kKT * kD) / 4; i += 256) {
      int idx = i * 4;
      int row = idx / kD, d = idx % kD;
      float4 f = *(const float4*)(Kg + (size_t)(kt + row) * kD + d);
      _Float16* p = &Ks[row * kKStr + d];
      p[0] = (_Float16)f.x; p[1] = (_Float16)f.y;
      p[2] = (_Float16)f.z; p[3] = (_Float16)f.w;
    }
    // ---- Stage V tile transposed: Vs[d][m] so B-frags are packed loads ----
#pragma unroll
    for (int i = tid; i < (kKT * kD) / 4; i += 256) {
      int idx = i * 4;
      int m = idx / kD, d = idx % kD;
      float4 f = *(const float4*)(Vg + (size_t)(kt + m) * kD + d);
      Vs[(d + 0) * kVStr + m] = (_Float16)f.x;
      Vs[(d + 1) * kVStr + m] = (_Float16)f.y;
      Vs[(d + 2) * kVStr + m] = (_Float16)f.z;
      Vs[(d + 3) * kVStr + m] = (_Float16)f.w;
    }
    // Warm L2 for the next tile while this one computes.
    if (kt + kKT < kN) {
      __builtin_prefetch(Kg + (size_t)(kt + kKT + (tid >> 3)) * kD + (tid & 7) * 8, 0, 1);
      __builtin_prefetch(Vg + (size_t)(kt + kKT + (tid >> 3)) * kD + (tid & 7) * 8, 0, 1);
    }
    wait_async0();          // mask bytes landed in LDS (ASYNCcnt not barrier-covered)
    __syncthreads();

    // ---- S^T = K * Q^T (f32 accum): two 16-key tiles, A = K rows from LDS ----
    v8f sacc[2];
#pragma unroll
    for (int nt = 0; nt < 2; ++nt) {
      F16Frag a0, a1;
#pragma unroll
      for (int v = 0; v < 8; ++v) {
        a0.u[v] = *(const unsigned int*)&Ks[(nt * 16 + lr) * kKStr + kmapA[v]];
        a1.u[v] = *(const unsigned int*)&Ks[(nt * 16 + lr) * kKStr + 32 + kmapA[v]];
      }
      v8f c = {};
      c = __builtin_amdgcn_wmma_f32_16x16x32_f16(false, a0.v, false, qb[0].v,
                                                 (short)0, c, false, false);
      c = __builtin_amdgcn_wmma_f32_16x16x32_f16(false, a1.v, false, qb[1].v,
                                                 (short)0, c, false, false);
      sacc[nt] = c;
    }

    // ---- tanh (native) + pack: S^T C-layout is exactly the A-fragment of
    //      O = S*V for this lane (keys {8lh..}+{16+8lh..}), no LDS bounce ----
    F16Frag sa;
#pragma unroll
    for (int v = 0; v < 8; ++v) {
      int nt = v >> 2, i = 2 * (v & 3);
      sa.h[2 * v + 0] = (_Float16)hw_tanhf(sacc[nt][i + 0]);
      sa.h[2 * v + 1] = (_Float16)hw_tanhf(sacc[nt][i + 1]);
    }

    // ---- Mask: this lane's mask row, two contiguous 8-byte runs ----
    {
      uint2 m0 = *(const uint2*)(mRowPtr + 8 * lh);        // keys 8lh..8lh+7
      uint2 m1 = *(const uint2*)(mRowPtr + 16 + 8 * lh);   // keys 16+8lh..+7
      sa.u[0] = mask_pair(sa.u[0], m0.x, 0x0C010C00u);
      sa.u[1] = mask_pair(sa.u[1], m0.x, 0x0C030C02u);
      sa.u[2] = mask_pair(sa.u[2], m0.y, 0x0C010C00u);
      sa.u[3] = mask_pair(sa.u[3], m0.y, 0x0C030C02u);
      sa.u[4] = mask_pair(sa.u[4], m1.x, 0x0C010C00u);
      sa.u[5] = mask_pair(sa.u[5], m1.x, 0x0C030C02u);
      sa.u[6] = mask_pair(sa.u[6], m1.y, 0x0C010C00u);
      sa.u[7] = mask_pair(sa.u[7], m1.y, 0x0C030C02u);
    }

    // ---- O += S * V over this key tile (K=32), 4 d-tiles ----
#pragma unroll
    for (int dt = 0; dt < 4; ++dt) {
      F16Frag bv;
#pragma unroll
      for (int v = 0; v < 8; ++v)
        bv.u[v] = *(const unsigned int*)&Vs[(dt * 16 + lr) * kVStr + kmapB[v]];
      oacc[dt] = __builtin_amdgcn_wmma_f32_16x16x32_f16(false, sa.v, false, bv.v,
                                                        (short)0, oacc[dt], false, false);
    }
    __syncthreads();
  }

  // ---- Write O (f32) from C/D layout; lanes 0..15 cover contiguous 16 cols ----
#pragma unroll
  for (int dt = 0; dt < 4; ++dt)
#pragma unroll
    for (int i = 0; i < 8; ++i) {
      int m = wave * 16 + i + 8 * lh;
      Og[(size_t)m * kD + dt * 16 + lr] = oacc[dt][i];
    }
}

extern "C" void kernel_launch(void* const* d_in, const int* in_sizes, int n_in,
                              void* d_out, int out_size, void* d_ws, size_t ws_size,
                              hipStream_t stream) {
  (void)in_sizes; (void)n_in; (void)out_size; (void)d_ws; (void)ws_size;
  const float*         q = (const float*)d_in[0];
  const float*         k = (const float*)d_in[1];
  const float*         v = (const float*)d_in[2];
  const unsigned char* m = (const unsigned char*)d_in[3];
  float*               o = (float*)d_out;

  dim3 grid(kN / kMTile, 4 * 8);   // (16 query tiles, B*H heads)
  dim3 block(256);
  attn_tanh_wmma<<<grid, block, 0, stream>>>(q, k, v, m, o);
}
